// BahdanauAttention_18399639896619
// MI455X (gfx1250) — matrix operand order repaired and checked by reading, not compiled, not run on hardware
//
#include <hip/hip_runtime.h>
#include <hip/hip_bf16.h>
#include <math.h>

typedef __attribute__((ext_vector_type(16))) __bf16 v16bf;
typedef __attribute__((ext_vector_type(8)))  __bf16 v8bf;
typedef __attribute__((ext_vector_type(4)))  __bf16 v4bf;
typedef __attribute__((ext_vector_type(8)))  float  v8f;

namespace {
constexpr int kH = 1024;
constexpr int kB = 32;
constexpr int kS = 2048;
constexpr int kTileS = 64;                 // key rows per workgroup = 4 x 16-row WMMA tiles
constexpr int kRowTiles = kTileS / 16;
constexpr int kLdsStride = kH + 8;         // bf16 elements, padded vs. bank conflicts
constexpr int kWaves = 4;
constexpr int kThreads = kWaves * 32;      // wave32
constexpr int kNSplit = 16;                // context partial splits over S
constexpr float kNegMax = -3.402823466e38f;
}

union AFrag { v16bf v; v8bf h[2]; };

// Branch-free tanh: exact at 0, saturates to +-1 for large |x| (exp -> inf -> 2/inf = 0).
__device__ __forceinline__ float fast_tanh(float x) {
  const float ax = fabsf(x);
  const float e  = __expf(ax + ax);
  const float r  = 1.0f - 2.0f / (e + 1.0f);
  return __builtin_copysignf(r, x);
}

// ---------------- Ua_w f32 -> bf16 (L2-resident B matrix) ----------------
__global__ __launch_bounds__(256) void bahdanau_cvt_bf16_kernel(
    const float* __restrict__ src, __bf16* __restrict__ dst) {
  const int i = blockIdx.x * 256 + threadIdx.x;       // over H*H/4
  const float4 f = ((const float4*)src)[i];
  v4bf o;
  o[0] = (__bf16)f.x; o[1] = (__bf16)f.y; o[2] = (__bf16)f.z; o[3] = (__bf16)f.w;
  ((v4bf*)dst)[i] = o;
}

// ---------------- q_proj = query @ Wa_w^T + Wa_b (tiny) ----------------
__global__ __launch_bounds__(256) void bahdanau_qproj_kernel(
    const float* __restrict__ query, const float* __restrict__ waW,
    const float* __restrict__ waB, float* __restrict__ qproj) {
  const int idx = blockIdx.x * 256 + threadIdx.x;     // over B*H
  const int b = idx >> 10;
  const int g = idx & (kH - 1);
  const float4* q = (const float4*)(query + (size_t)b * kH);
  const float4* w = (const float4*)(waW + (size_t)g * kH);
  float acc = 0.0f;
  #pragma unroll 4
  for (int i = 0; i < kH / 4; ++i) {
    const float4 qv = q[i];
    const float4 wv = w[i];
    acc += qv.x * wv.x + qv.y * wv.y + qv.z * wv.z + qv.w * wv.w;
  }
  qproj[idx] = acc + waB[g];
}

// ---- fused: scores[b,s] = sum_g tanh(qproj[b,g]+Ua_b[g]+ (keys@Ua^T)[s,g]) * Va[g] ----
__global__ __launch_bounds__(kThreads) void bahdanau_scores_kernel(
    const float* __restrict__ keys, const __bf16* __restrict__ uaBf,
    const float* __restrict__ qproj, const float* __restrict__ uaB,
    const float* __restrict__ vaW, const float* __restrict__ vaB,
    float* __restrict__ scores) {
  extern __shared__ __align__(16) char smem[];
  __bf16* ldsK = (__bf16*)smem;
  float* ldsScore = (float*)(smem + (size_t)kTileS * kLdsStride * sizeof(__bf16));

  const int b    = blockIdx.y;
  const int s0   = blockIdx.x * kTileS;
  const int tid  = threadIdx.x;
  const int lane = tid & 31;
  const int wave = tid >> 5;
  const int lrow = lane & 15;   // A row within tile / C column (N)
  const int half = lane >> 4;   // which K-half (A/B) / which M-half (C)

  if (tid < kTileS) ldsScore[tid] = 0.0f;

  // Stage 64x1024 f32 keys tile as bf16 into LDS (read keys exactly once here).
  const float* gk = keys + ((size_t)b * kS + s0) * kH;
  for (int i = tid; i < kTileS * (kH / 4); i += kThreads) {
    const int row = i >> 8;                 // kH/4 == 256
    const int c4  = i & 255;
    const float4 f = ((const float4*)(gk + (size_t)row * kH))[c4];
    v4bf o;
    o[0] = (__bf16)f.x; o[1] = (__bf16)f.y; o[2] = (__bf16)f.z; o[3] = (__bf16)f.w;
    *(v4bf*)(ldsK + row * kLdsStride + c4 * 4) = o;
  }
  __syncthreads();

  float rowAcc[kRowTiles][8];
  #pragma unroll
  for (int t = 0; t < kRowTiles; ++t)
    #pragma unroll
    for (int m = 0; m < 8; ++m) rowAcc[t][m] = 0.0f;

  // Per-lane base pointers (ISA 16-bit WMMA fragment layouts).
  const __bf16* abase = ldsK + lrow * kLdsStride + half * 8;

  // Each wave owns a strided set of PAIRS of 16-column output blocks.
  // NB=2 columns per iteration: 8 independent WMMA chains, A frags reused 2x.
  #pragma unroll 1
  for (int nbp = wave; nbp < kH / 32; nbp += kWaves) {
    const int g0 = (nbp * 2) * 16 + lrow;          // output feature columns
    const int g1 = g0 + 16;
    const float qp0 = qproj[b * kH + g0] + uaB[g0];
    const float qp1 = qproj[b * kH + g1] + uaB[g1];
    const float va0 = vaW[g0];
    const float va1 = vaW[g1];

    const __bf16* bbase0 = uaBf + (size_t)g0 * kH + half * 8;
    const __bf16* bbase1 = uaBf + (size_t)g1 * kH + half * 8;

    v8f acc0[kRowTiles] = {};
    v8f acc1[kRowTiles] = {};

    // Double-buffered fragments: load kc+1 while computing kc.
    AFrag aA[kRowTiles], aB[kRowTiles];
    AFrag b0A, b1A, b0B, b1B;

    // Preload kc = 0 into buffer A.
    b0A.h[0] = *(const v8bf*)(bbase0);
    b0A.h[1] = *(const v8bf*)(bbase0 + 16);
    b1A.h[0] = *(const v8bf*)(bbase1);
    b1A.h[1] = *(const v8bf*)(bbase1 + 16);
    #pragma unroll
    for (int t = 0; t < kRowTiles; ++t) {
      const __bf16* ar = abase + t * 16 * kLdsStride;
      aA[t].h[0] = *(const v8bf*)(ar);
      aA[t].h[1] = *(const v8bf*)(ar + 16);
    }

    #pragma unroll 1
    for (int kc = 0; kc < kH / 32; kc += 2) {
      // ---- prefetch kc+1 into buffer B ----
      {
        const int kn = (kc + 1) * 32;
        b0B.h[0] = *(const v8bf*)(bbase0 + kn);
        b0B.h[1] = *(const v8bf*)(bbase0 + kn + 16);
        b1B.h[0] = *(const v8bf*)(bbase1 + kn);
        b1B.h[1] = *(const v8bf*)(bbase1 + kn + 16);
        #pragma unroll
        for (int t = 0; t < kRowTiles; ++t) {
          const __bf16* ar = abase + t * 16 * kLdsStride + kn;
          aB[t].h[0] = *(const v8bf*)(ar);
          aB[t].h[1] = *(const v8bf*)(ar + 16);
        }
      }
      // ---- compute kc with buffer A ----
      #pragma unroll
      for (int t = 0; t < kRowTiles; ++t) {
        acc0[t] = __builtin_amdgcn_wmma_f32_16x16x32_bf16(
            false, aA[t].v, false, b0A.v, (short)0, acc0[t], false, false);
        acc1[t] = __builtin_amdgcn_wmma_f32_16x16x32_bf16(
            false, aA[t].v, false, b1A.v, (short)0, acc1[t], false, false);
      }
      // ---- prefetch kc+2 into buffer A (dummy re-read on final iteration) ----
      {
        const int kn = ((kc + 2 < kH / 32) ? (kc + 2) : 0) * 32;
        b0A.h[0] = *(const v8bf*)(bbase0 + kn);
        b0A.h[1] = *(const v8bf*)(bbase0 + kn + 16);
        b1A.h[0] = *(const v8bf*)(bbase1 + kn);
        b1A.h[1] = *(const v8bf*)(bbase1 + kn + 16);
        #pragma unroll
        for (int t = 0; t < kRowTiles; ++t) {
          const __bf16* ar = abase + t * 16 * kLdsStride + kn;
          aA[t].h[0] = *(const v8bf*)(ar);
          aA[t].h[1] = *(const v8bf*)(ar + 16);
        }
      }
      // ---- compute kc+1 with buffer B ----
      #pragma unroll
      for (int t = 0; t < kRowTiles; ++t) {
        acc0[t] = __builtin_amdgcn_wmma_f32_16x16x32_bf16(
            false, aB[t].v, false, b0B.v, (short)0, acc0[t], false, false);
        acc1[t] = __builtin_amdgcn_wmma_f32_16x16x32_bf16(
            false, aB[t].v, false, b1B.v, (short)0, acc1[t], false, false);
      }
    }

    // Fused epilogue: tanh + Va dot, accumulated per lane across column blocks.
    #pragma unroll
    for (int t = 0; t < kRowTiles; ++t)
      #pragma unroll
      for (int m = 0; m < 8; ++m)
        rowAcc[t][m] += fast_tanh(qp0 + acc0[t][m]) * va0 +
                        fast_tanh(qp1 + acc1[t][m]) * va1;
  }

  // Reduce over the 16 columns held by each half-wave, then combine waves in LDS.
  #pragma unroll
  for (int t = 0; t < kRowTiles; ++t) {
    #pragma unroll
    for (int m = 0; m < 8; ++m) {
      float v = rowAcc[t][m];
      v += __shfl_xor(v, 1, 32);
      v += __shfl_xor(v, 2, 32);
      v += __shfl_xor(v, 4, 32);
      v += __shfl_xor(v, 8, 32);
      if (lrow == 0) atomicAdd(&ldsScore[t * 16 + half * 8 + m], v);
    }
  }
  __syncthreads();
  if (tid < kTileS)
    scores[(size_t)b * kS + s0 + tid] = ldsScore[tid] + vaB[0];
}

// ---------------- masked softmax over S per batch row ----------------
__global__ __launch_bounds__(256) void bahdanau_softmax_kernel(
    const float* __restrict__ scores, const int* __restrict__ mask,
    float* __restrict__ weights) {
  constexpr int PER = kS / 256;   // 8
  const int b = blockIdx.x;
  const int tid = threadIdx.x;
  __shared__ float red[8];

  float vals[PER];
  float mx = kNegMax;
  #pragma unroll
  for (int i = 0; i < PER; ++i) {
    const int s = tid + i * 256;
    float sc = scores[(size_t)b * kS + s];
    if (mask[(size_t)b * kS + s] == 0) sc = kNegMax;
    vals[i] = sc;
    mx = fmaxf(mx, sc);
  }
  #pragma unroll
  for (int o = 16; o > 0; o >>= 1) mx = fmaxf(mx, __shfl_xor(mx, o, 32));
  if ((tid & 31) == 0) red[tid >> 5] = mx;
  __syncthreads();
  mx = red[0];
  #pragma unroll
  for (int w = 1; w < 8; ++w) mx = fmaxf(mx, red[w]);
  __syncthreads();

  float sum = 0.0f;
  #pragma unroll
  for (int i = 0; i < PER; ++i) {
    const float e = __expf(vals[i] - mx);
    vals[i] = e;
    sum += e;
  }
  #pragma unroll
  for (int o = 16; o > 0; o >>= 1) sum += __shfl_xor(sum, o, 32);
  if ((tid & 31) == 0) red[tid >> 5] = sum;
  __syncthreads();
  sum = red[0];
  #pragma unroll
  for (int w = 1; w < 8; ++w) sum += red[w];

  const float inv = 1.0f / sum;
  #pragma unroll
  for (int i = 0; i < PER; ++i)
    weights[(size_t)b * kS + tid + i * 256] = vals[i] * inv;
}

// ---------------- context = weights . keys (split + deterministic reduce) ----------------
__global__ __launch_bounds__(256) void bahdanau_context_partial_kernel(
    const float* __restrict__ keys, const float* __restrict__ weights,
    float* __restrict__ partial) {
  constexpr int SCHUNK = kS / kNSplit;   // 128
  const int split = blockIdx.x;
  const int b = blockIdx.y;
  const int tid = threadIdx.x;
  __shared__ float wl[SCHUNK];
  if (tid < SCHUNK) wl[tid] = weights[(size_t)b * kS + split * SCHUNK + tid];
  __syncthreads();
  const float* kb = keys + ((size_t)b * kS + (size_t)split * SCHUNK) * kH;
  float4 acc = {0.0f, 0.0f, 0.0f, 0.0f};
  for (int s = 0; s < SCHUNK; ++s) {
    const float w = wl[s];
    const float4 k = ((const float4*)(kb + (size_t)s * kH))[tid];
    acc.x += w * k.x; acc.y += w * k.y; acc.z += w * k.z; acc.w += w * k.w;
  }
  ((float4*)(partial + ((size_t)(b * kNSplit + split)) * kH))[tid] = acc;
}

__global__ __launch_bounds__(256) void bahdanau_context_reduce_kernel(
    const float* __restrict__ partial, float* __restrict__ context) {
  const int idx = blockIdx.x * 256 + threadIdx.x;   // over B*H
  const int b = idx >> 10;
  const int h = idx & (kH - 1);
  float acc = 0.0f;
  #pragma unroll
  for (int k = 0; k < kNSplit; ++k)
    acc += partial[((size_t)(b * kNSplit + k)) * kH + h];
  context[idx] = acc;
}

extern "C" void kernel_launch(void* const* d_in, const int* in_sizes, int n_in,
                              void* d_out, int out_size, void* d_ws, size_t ws_size,
                              hipStream_t stream) {
  (void)in_sizes; (void)n_in; (void)out_size; (void)ws_size;
  const float* query = (const float*)d_in[0];
  const float* keys  = (const float*)d_in[1];
  const int*   mask  = (const int*)d_in[2];
  const float* waW   = (const float*)d_in[3];
  const float* waB   = (const float*)d_in[4];
  const float* uaW   = (const float*)d_in[5];
  const float* uaB   = (const float*)d_in[6];
  const float* vaW   = (const float*)d_in[7];
  const float* vaB   = (const float*)d_in[8];

  float* out = (float*)d_out;
  float* context = out;                 // [B,H]
  float* weights = out + kB * kH;       // [B,S]

  char* ws = (char*)d_ws;
  __bf16* uaBf  = (__bf16*)ws;          ws += (size_t)kH * kH * sizeof(__bf16);
  float* qproj  = (float*)ws;           ws += (size_t)kB * kH * sizeof(float);
  float* scores = (float*)ws;           ws += (size_t)kB * kS * sizeof(float);
  float* partial = (float*)ws;          // [B, kNSplit, H]

  bahdanau_cvt_bf16_kernel<<<kH * kH / 4 / 256, 256, 0, stream>>>(uaW, uaBf);
  bahdanau_qproj_kernel<<<kB * kH / 256, 256, 0, stream>>>(query, waW, waB, qproj);

  constexpr size_t kScoresSmem =
      (size_t)kTileS * kLdsStride * sizeof(__bf16) + kTileS * sizeof(float);
  bahdanau_scores_kernel<<<dim3(kS / kTileS, kB), kThreads, kScoresSmem, stream>>>(
      keys, uaBf, qproj, uaB, vaW, vaB, scores);

  bahdanau_softmax_kernel<<<kB, 256, 0, stream>>>(scores, mask, weights);
  bahdanau_context_partial_kernel<<<dim3(kNSplit, kB), 256, 0, stream>>>(
      keys, weights, partial);
  bahdanau_context_reduce_kernel<<<kB * kH / 256, 256, 0, stream>>>(partial, context);
}
